// Net_14422500180486
// MI455X (gfx1250) — matrix operand order repaired and checked by reading, not compiled, not run on hardware
//
#include <hip/hip_runtime.h>
#include <hip/hip_bf16.h>

typedef __attribute__((ext_vector_type(16))) _Float16 v16h;
typedef __attribute__((ext_vector_type(8)))  _Float16 h8;
typedef __attribute__((ext_vector_type(8)))  float    v8f;

#define B_  32
#define H_  128
#define G_  512   // 4*H
#define D0_ 64
#define RING 64          // h0 pipeline ring slots (power of 2)
#define RMASK (RING - 1)

union V16u { v16h v; h8 h[2]; };

__device__ __forceinline__ float sigm(float x) {
    return __builtin_amdgcn_rcpf(1.0f + __expf(-x));
}
__device__ __forceinline__ float tanh_fast(float x) {
    return 2.0f * __builtin_amdgcn_rcpf(1.0f + __expf(-2.0f * x)) - 1.0f;
}
__device__ __forceinline__ v8f splat8(float x) {
    v8f v;
#pragma unroll
    for (int r = 0; r < 8; ++r) v[r] = x;
    return v;
}
__device__ __forceinline__ int aload_acq(int* p) {
    return __hip_atomic_load(p, __ATOMIC_ACQUIRE, __HIP_MEMORY_SCOPE_AGENT);
}
__device__ __forceinline__ void astore_rel(int* p, int v) {
    __hip_atomic_store(p, v, __ATOMIC_RELEASE, __HIP_MEMORY_SCOPE_AGENT);
}
__device__ __forceinline__ void astore_rlx(int* p, int v) {
    __hip_atomic_store(p, v, __ATOMIC_RELAXED, __HIP_MEMORY_SCOPE_AGENT);
}

// ---------------------------------------------------------------------------
// Weight repack: fp32 [4H,D]+[4H,H] -> fp16 [512][KP] fused [Wih|Whh], padded
// rows for LDS bank staggering; bias = bih + bhh (fp32). Also zero-initializes
// the pipeline counters every launch (stream-ordered before the scan kernel,
// so replays and 0xAA workspace poison are both handled).
// ---------------------------------------------------------------------------
__global__ void lstm_prep(const float* __restrict__ Wih0, const float* __restrict__ Whh0,
                          const float* __restrict__ bih0, const float* __restrict__ bhh0,
                          const float* __restrict__ Wih1, const float* __restrict__ Whh1,
                          const float* __restrict__ bih1, const float* __restrict__ bhh1,
                          _Float16* __restrict__ W0p, _Float16* __restrict__ W1p,
                          float* __restrict__ b0, float* __restrict__ b1,
                          int* __restrict__ produced, int* __restrict__ consumed) {
    const int N0 = G_ * 200;            // layer0: K=192 (+8 pad)
    const int N1 = G_ * 264;            // layer1: K=256 (+8 pad)
    const int total = N0 + N1 + 2 * G_;
    for (int idx = blockIdx.x * blockDim.x + threadIdx.x; idx < total;
         idx += gridDim.x * blockDim.x) {
        if (idx == 0) { produced[0] = 0; consumed[0] = 0; }
        if (idx < N0) {
            int n = idx / 200, k = idx % 200;
            float v = 0.0f;
            if (k < 64)        v = Wih0[n * 64 + k];
            else if (k < 192)  v = Whh0[n * 128 + (k - 64)];
            W0p[idx] = (_Float16)v;
        } else if (idx < N0 + N1) {
            int j = idx - N0;
            int n = j / 264, k = j % 264;
            float v = 0.0f;
            if (k < 128)       v = Wih1[n * 128 + k];
            else if (k < 256)  v = Whh1[n * 128 + (k - 128)];
            W1p[j] = (_Float16)v;
        } else {
            int j = idx - N0 - N1;
            if (j < G_) b0[j] = bih0[j] + bhh0[j];
            else        b1[j - G_] = bih1[j - G_] + bhh1[j - G_];
        }
    }
}

// ---------------------------------------------------------------------------
// One LSTM layer as a persistent single-workgroup scan.
// FIRST=true : layer0, reads fp32 x [B,T,64], produces fp16 h0 into ring,
//              publishes `produced` (release) after each slot is written.
// FIRST=false: layer1, consumes ring slots (acquire on `produced`), writes
//              fp32 output [B,T,H], publishes `consumed` for back-pressure.
// 16 waves: wave = (mt 0..1) x (ht 0..7); each wave owns the 4 gate tiles
// (i,f,g,o) for its 16 batch rows x 16 h columns; c lives in v8f registers.
// ---------------------------------------------------------------------------
template <int KIN, int KP, bool FIRST>
__device__ void scan_layer(const float* __restrict__ xin,
                           const _Float16* __restrict__ ring_r,
                           const _Float16* __restrict__ Wp,
                           const float* __restrict__ bias,
                           _Float16* __restrict__ ring_w,
                           float* __restrict__ out,
                           int* produced, int* consumed,
                           _Float16* Wl, _Float16* xh, int T) {
    constexpr int K = KIN + H_;
    constexpr int KITER = K / 32;

    const int tid   = threadIdx.x;
    const int lane  = tid & 31;
    const int wv    = tid >> 5;         // 0..15
    const int mt    = wv >> 3;          // 0..1  (batch tile)
    const int ht    = wv & 7;           // 0..7  (h-column tile)
    const int l16   = lane & 15;
    const int khalf = lane >> 4;        // 0/1

    // ---- one-time: weights -> LDS, zero xh (h state starts at 0) ----
    {
        const uint4* src = (const uint4*)Wp;
        uint4* dst = (uint4*)Wl;
        const int n16 = (G_ * KP) / 8;
        for (int i = tid; i < n16; i += 512) dst[i] = src[i];
        uint4 z; z.x = z.y = z.z = z.w = 0u;
        uint4* xz = (uint4*)xh;
        const int nz = (B_ * KP) / 8;
        for (int i = tid; i < nz; i += 512) xz[i] = z;
    }

    float bgv[4];
#pragma unroll
    for (int g = 0; g < 4; ++g) bgv[g] = bias[g * H_ + ht * 16 + l16];

    v8f cc = {};                        // persistent cell state tile

    int avail = 0;                      // consumer: cached `produced`
    int cons  = 0;                      // producer: cached `consumed`

    // preload step-0 input
    float4 xcur4;
    uint4  hcur;
    const int sb = tid >> 4, sq = tid & 15;
    if (FIRST) {
        xcur4 = *(const float4*)(xin + ((size_t)sb * T) * D0_ + sq * 4);
    } else {
        while (avail < 1) {
            avail = aload_acq(produced);
            if (avail < 1) __builtin_amdgcn_s_sleep(1);
        }
        hcur = ((const uint4*)ring_r)[tid];   // slot 0
    }

    __syncthreads();

    const int aoff = (mt * 16 + l16) * KP;    // A-operand row base (batch row)

    for (int t = 0; t < T; ++t) {
        // ---- stage input tile into LDS (x slots [0,KIN)) ----
        if (FIRST) {
            union { _Float16 h[4]; uint2 u; } pk;
            pk.h[0] = (_Float16)xcur4.x;
            pk.h[1] = (_Float16)xcur4.y;
            pk.h[2] = (_Float16)xcur4.z;
            pk.h[3] = (_Float16)xcur4.w;
            *(uint2*)&xh[sb * KP + sq * 4] = pk.u;
        } else {
            *(uint4*)&xh[sb * KP + sq * 8] = hcur;
        }
        __syncthreads();

        // ---- pipeline bookkeeping ----
        if (FIRST) {
            // all threads fenced ring-slot (t-1) before this barrier
            if (tid == 0 && t > 0) astore_rel(produced, t);
        } else {
            if (tid == 0) astore_rlx(consumed, t + 1);   // slot t fully read
        }

        // ---- prefetch next step's input (hidden under the GEMM) ----
        if (t + 1 < T) {
            if (FIRST) {
                xcur4 = *(const float4*)(xin + ((size_t)sb * T + (t + 1)) * D0_ + sq * 4);
            } else {
                while (avail < t + 2) {      // cached; rarely hits memory
                    avail = aload_acq(produced);
                    if (avail < t + 2) __builtin_amdgcn_s_sleep(1);
                }
                hcur = ((const uint4*)(ring_r + (size_t)((t + 1) & RMASK) * (B_ * H_)))[tid];
            }
        }

        // ---- gate GEMM: 4 gate tiles, K accumulated via WMMA f16->f32 ----
        v8f acc0 = splat8(bgv[0]);
        v8f acc1 = splat8(bgv[1]);
        v8f acc2 = splat8(bgv[2]);
        v8f acc3 = splat8(bgv[3]);
#pragma unroll
        for (int ki = 0; ki < KITER; ++ki) {
            const int kb = ki * 32;
            // A operand (16x32 f16): lane l16 = row, khalf selects K-halves
            V16u a;
            a.h[0] = *(const h8*)&xh[aoff + kb + khalf * 8];
            a.h[1] = *(const h8*)&xh[aoff + kb + 16 + khalf * 8];
            // B operands (32x16 f16): lane l16 = column, khalf selects K-halves
            const int kw = kb + khalf * 8;
            V16u bm[4];
#pragma unroll
            for (int g = 0; g < 4; ++g) {
                const int n = g * H_ + ht * 16 + l16;
                bm[g].h[0] = *(const h8*)&Wl[n * KP + kw];
                bm[g].h[1] = *(const h8*)&Wl[n * KP + kw + 16];
            }
            acc0 = __builtin_amdgcn_wmma_f32_16x16x32_f16(false, a.v, false, bm[0].v, (short)0, acc0, false, false);
            acc1 = __builtin_amdgcn_wmma_f32_16x16x32_f16(false, a.v, false, bm[1].v, (short)0, acc1, false, false);
            acc2 = __builtin_amdgcn_wmma_f32_16x16x32_f16(false, a.v, false, bm[2].v, (short)0, acc2, false, false);
            acc3 = __builtin_amdgcn_wmma_f32_16x16x32_f16(false, a.v, false, bm[3].v, (short)0, acc3, false, false);
        }
        __syncthreads();   // all xh reads done before h-slot rewrite

        // ---- activations + cell update (in-register, D-tile layout) ----
#pragma unroll
        for (int r = 0; r < 8; ++r) {
            const float iv = sigm(acc0[r]);
            const float fv = sigm(acc1[r]);
            const float gv = tanh_fast(acc2[r]);
            const float ov = sigm(acc3[r]);
            const float c2 = fv * cc[r] + iv * gv;
            cc[r] = c2;
            const float hv = ov * tanh_fast(c2);
            const int row = mt * 16 + (khalf ? (r + 8) : r);   // batch row
            const int col = ht * 16 + l16;                      // h index
            xh[row * KP + KIN + col] = (_Float16)hv;            // h for t+1
            if (FIRST) {
                ring_w[(size_t)(t & RMASK) * (B_ * H_) + row * H_ + col] = (_Float16)hv;
            } else {
                out[(size_t)row * T * H_ + (size_t)t * H_ + col] = hv;
            }
        }

        if (FIRST) {
            // back-pressure: next iteration overwrites slot (t+1-RING)
            if (t + 1 < T && cons < (t + 2) - RING) {
                while (cons < (t + 2) - RING) {
                    cons = aload_acq(consumed);
                    if (cons < (t + 2) - RING) __builtin_amdgcn_s_sleep(1);
                }
            }
            __threadfence();   // ring-slot t stores visible at device scope
        }
    }

    if (FIRST) {
        __threadfence();
        __syncthreads();
        if (tid == 0) astore_rel(produced, T);   // final slot published
    }
}

__global__ __launch_bounds__(512, 1)
void lstm_pipe(const float* __restrict__ x,
               const _Float16* __restrict__ W0p, const float* __restrict__ bias0,
               const _Float16* __restrict__ W1p, const float* __restrict__ bias1,
               _Float16* __restrict__ hring, int* produced, int* consumed,
               float* __restrict__ out, int T) {
    __shared__ _Float16 Wl[G_ * 264];   // sized for the larger layer
    __shared__ _Float16 xh[B_ * 264];
    if (blockIdx.x == 0) {
        scan_layer<64, 200, true>(x, nullptr, W0p, bias0, hring, nullptr,
                                  produced, consumed, Wl, xh, T);
    } else {
        scan_layer<128, 264, false>(nullptr, hring, W1p, bias1, nullptr, out,
                                    produced, consumed, Wl, xh, T);
    }
}

extern "C" void kernel_launch(void* const* d_in, const int* in_sizes, int n_in,
                              void* d_out, int out_size, void* d_ws, size_t ws_size,
                              hipStream_t stream) {
    const float* x    = (const float*)d_in[0];
    const float* Wih0 = (const float*)d_in[1];
    const float* Whh0 = (const float*)d_in[2];
    const float* bih0 = (const float*)d_in[3];
    const float* bhh0 = (const float*)d_in[4];
    const float* Wih1 = (const float*)d_in[5];
    const float* Whh1 = (const float*)d_in[6];
    const float* bih1 = (const float*)d_in[7];
    const float* bhh1 = (const float*)d_in[8];
    (void)n_in; (void)ws_size; (void)out_size;

    const int T = in_sizes[0] / (B_ * D0_);

    char* ws = (char*)d_ws;
    const size_t szW0 = (size_t)G_ * 200 * sizeof(_Float16);   // 204800
    const size_t szW1 = (size_t)G_ * 264 * sizeof(_Float16);   // 270336
    _Float16* W0p = (_Float16*)(ws);
    _Float16* W1p = (_Float16*)(ws + szW0);
    float*    b0  = (float*)(ws + szW0 + szW1);
    float*    b1  = (float*)(ws + szW0 + szW1 + G_ * sizeof(float));
    size_t off = szW0 + szW1 + 2 * G_ * sizeof(float);          // 479232
    int* produced = (int*)(ws + off);
    int* consumed = (int*)(ws + off + 128);
    off = (off + 1024 + 1023) & ~(size_t)1023;                  // 480256
    _Float16* hring = (_Float16*)(ws + off);                    // 64*B*H f16 = 512KB

    const int prepN = G_ * 200 + G_ * 264 + 2 * G_;
    lstm_prep<<<(prepN + 255) / 256, 256, 0, stream>>>(
        Wih0, Whh0, bih0, bhh0, Wih1, Whh1, bih1, bhh1,
        W0p, W1p, b0, b1, produced, consumed);

    lstm_pipe<<<2, 512, 0, stream>>>(x, W0p, b0, W1p, b1, hring,
                                     produced, consumed, (float*)d_out, T);
}
